// SSIMLoss_82094004895839
// MI455X (gfx1250) — matrix operand order repaired and checked by reading, not compiled, not run on hardware
//
#include <hip/hip_runtime.h>
#include <hip/hip_bf16.h>

// ---------------------------------------------------------------------------
// SSIM loss, fused, WMMA-based separable Gaussian convolution for gfx1250.
//   img: [32, 1, 1024, 1024] f32.  Window 11, sigma 1.5, zero ("same") pad.
//   Per 16x16 output tile:  Mid = In(32x32) * H(32x16)   (2x wmma bf16)
//                           Out = V(16x32) * Mid(32x16)  (1x wmma bf16)
//   5 conv fields (x1, x2, x1*x2, x1^2, x2^2) -> 15 WMMAs / tile, f32 accum.
//   Band matrices H/V are compile-time constant tables in WMMA operand
//   layout (packed bf16 dwords per (lane, dword)): 2x b128 loads, no VALU.
// ---------------------------------------------------------------------------

typedef __attribute__((ext_vector_type(16))) __bf16        v16bf;
typedef __attribute__((ext_vector_type(2)))  __bf16        v2bf;
typedef __attribute__((ext_vector_type(8)))  float         v8f;
typedef __attribute__((ext_vector_type(4)))  float         f32x4;
typedef __attribute__((ext_vector_type(8)))  unsigned int  v8u;

#define IMG_H 1024
#define IMG_W 1024
#define N_PIX (33554432.0)   // 32*1024*1024

// bf16(RNE) bit patterns of the normalized 11-tap Gaussian (sigma=1.5).
struct BandTables {
    unsigned int hb[32][8];   // H (32x16) in B-operand layout
    unsigned int va[32][8];   // V (16x32) in A-operand layout
};

constexpr BandTables make_tables() {
    // taps 0..10 as bf16 bits
    unsigned short b[11] = { 0x3A87, 0x3BF9, 0x3D13, 0x3DE0, 0x3E5A,
                             0x3E88,
                             0x3E5A, 0x3DE0, 0x3D13, 0x3BF9, 0x3A87 };
    BandTables t{};
    for (int lane = 0; lane < 32; ++lane) {
        const int m  = lane & 15;
        const int hi = lane >> 4;
        for (int c = 0; c < 8; ++c) {
            // H: element i -> K = (i<8 ? i : i+8) + 8*hi, N = m ; H[k][n] = w[k-n-3]
            unsigned int lo = 0, hh = 0;
            {
                const int k0 = ((2 * c) < 8 ? (2 * c) : (2 * c) + 8) + 8 * hi;
                const int d0 = k0 - m - 3;
                const int d1 = k0 + 1 - m - 3;
                lo = (d0 >= 0 && d0 <= 10) ? b[d0] : 0u;
                hh = (d1 >= 0 && d1 <= 10) ? b[d1] : 0u;
            }
            t.hb[lane][c] = lo | (hh << 16);
            // V: element i -> K = i + 16*hi, M = m ; V[mm][k] = w[k-mm]
            {
                const int i0 = 2 * c;
                const int k0 = i0 + 16 * hi;
                const int k1 = k0 + 1;   // i0 and i0+1 are in the same chunk
                const int d0 = k0 - m;
                const int d1 = k1 - m;
                const unsigned int l2 = (d0 >= 0 && d0 <= 10) ? b[d0] : 0u;
                const unsigned int h2 = (d1 >= 0 && d1 <= 10) ? b[d1] : 0u;
                t.va[lane][c] = l2 | (h2 << 16);
            }
        }
    }
    return t;
}

__device__ __constant__ BandTables g_tab = make_tables();

__device__ __forceinline__ float gload(const float* p, int r, int c) {
    return (r >= 0 && r < IMG_H && c >= 0 && c < IMG_W) ? p[r * IMG_W + c] : 0.0f;
}

// Pack two f32 into one bf16x2 dword (lowers to v_cvt_pk_bf16_f32).
__device__ __forceinline__ unsigned int pack2(float lo, float hi) {
    v2bf p;
    p[0] = (__bf16)lo;
    p[1] = (__bf16)hi;
    return __builtin_bit_cast(unsigned int, p);
}

// Elementwise bf16 product of two WMMA operand vectors: 8x v_pk_mul_bf16.
__device__ __forceinline__ v16bf pkmul_bf16(v16bf a, v16bf b) {
    v8u ua = __builtin_bit_cast(v8u, a);
    v8u ub = __builtin_bit_cast(v8u, b);
    v8u ud;
#pragma unroll
    for (int c = 0; c < 8; ++c) {
        unsigned int t;
        asm("v_pk_mul_bf16 %0, %1, %2" : "=v"(t) : "v"(ua[c]), "v"(ub[c]));
        ud[c] = t;
    }
    return __builtin_bit_cast(v16bf, ud);
}

__global__ void ssim_init_kernel(double* acc) {
    if (threadIdx.x < 64) acc[threadIdx.x] = 0.0;
}

__global__ void __launch_bounds__(256)
ssim_main_kernel(const float* __restrict__ img1,
                 const float* __restrict__ img2,
                 double* __restrict__ acc) {
    const float C1 = 0.0001f;      // 0.01^2
    const float C2 = 0.0009f;      // 0.03^2

    // Wave-private packed-bf16 Mid scratch: 16 dword-rows (K-pairs) x 16 cols,
    // row stride 20 dwords => conflict-free for both access patterns.
    __shared__ unsigned int midp[8][16 * 20];
    __shared__ float wpart[8];

    const int wave = threadIdx.x >> 5;
    const int lane = threadIdx.x & 31;
    const int m    = lane & 15;         // row (A) / col N (B,C,D)
    const int hi   = lane >> 4;         // lane-group selector

    // tile id: 64x64 tiles per image, 32 images, 8 tiles (waves) per block
    const int tile = blockIdx.x * 8 + wave;
    const int tpi  = 64 * 64;
    const int img  = tile / tpi;
    const int t2   = tile - img * tpi;
    const int tr   = t2 >> 6;
    const int tc   = t2 & 63;
    const int r0   = tr * 16;
    const int c0   = tc * 16;

    const float* b1 = img1 + (size_t)img * IMG_H * IMG_W;
    const float* b2 = img2 + (size_t)img * IMG_H * IMG_W;

    // Band matrices in operand layout: 2x b128 per table, no VALU.
    v8u hu, vu;
#pragma unroll
    for (int c = 0; c < 8; ++c) hu[c] = g_tab.hb[lane][c];
#pragma unroll
    for (int c = 0; c < 8; ++c) vu[c] = g_tab.va[lane][c];
    const v16bf Hb = __builtin_bit_cast(v16bf, hu);
    const v16bf Va = __builtin_bit_cast(v16bf, vu);

    // Input region: rows r0-5 .. r0+26, cols c0-8 .. c0+23 (32B-aligned chunks)
    const int row_base = r0 - 5;
    const int col_base = c0 - 8;
    const bool interior = (r0 >= 5) && (r0 + 26 < IMG_H) &&
                          (c0 >= 8) && (c0 + 23 < IMG_W);

    // Raw halo in WMMA 16-bit A-layout order, packed to bf16 dwords:
    //   element i (i<8): K = i + 8*hi ;  (i>=8): K = i + 8 + 8*hi
    //   row = row_base + 16*t + m
    v16bf ab1[2], ab2[2];
    if (interior) {
#pragma unroll
        for (int t = 0; t < 2; ++t) {
            const int row = row_base + t * 16 + m;
            const float* rp1 = b1 + (size_t)row * IMG_W + col_base + hi * 8;
            const float* rp2 = b2 + (size_t)row * IMG_W + col_base + hi * 8;
            f32x4 u0 = *(const f32x4*)(rp1 + 0);
            f32x4 u1 = *(const f32x4*)(rp1 + 4);
            f32x4 u2 = *(const f32x4*)(rp1 + 16);
            f32x4 u3 = *(const f32x4*)(rp1 + 20);
            f32x4 v0 = *(const f32x4*)(rp2 + 0);
            f32x4 v1 = *(const f32x4*)(rp2 + 4);
            f32x4 v2 = *(const f32x4*)(rp2 + 16);
            f32x4 v3 = *(const f32x4*)(rp2 + 20);
            v8u w1, w2;
            w1[0] = pack2(u0[0], u0[1]);  w1[1] = pack2(u0[2], u0[3]);
            w1[2] = pack2(u1[0], u1[1]);  w1[3] = pack2(u1[2], u1[3]);
            w1[4] = pack2(u2[0], u2[1]);  w1[5] = pack2(u2[2], u2[3]);
            w1[6] = pack2(u3[0], u3[1]);  w1[7] = pack2(u3[2], u3[3]);
            w2[0] = pack2(v0[0], v0[1]);  w2[1] = pack2(v0[2], v0[3]);
            w2[2] = pack2(v1[0], v1[1]);  w2[3] = pack2(v1[2], v1[3]);
            w2[4] = pack2(v2[0], v2[1]);  w2[5] = pack2(v2[2], v2[3]);
            w2[6] = pack2(v3[0], v3[1]);  w2[7] = pack2(v3[2], v3[3]);
            ab1[t] = __builtin_bit_cast(v16bf, w1);
            ab2[t] = __builtin_bit_cast(v16bf, w2);
        }
    } else {
#pragma unroll
        for (int t = 0; t < 2; ++t) {
            const int row = row_base + t * 16 + m;
            float t1[16], t3[16];
#pragma unroll
            for (int i = 0; i < 16; ++i) {
                const int k = (i < 8 ? i : i + 8) + 8 * hi;
                const int col = col_base + k;
                t1[i] = gload(b1, row, col);
                t3[i] = gload(b2, row, col);
            }
            v8u w1, w2;
#pragma unroll
            for (int c = 0; c < 8; ++c) {
                w1[c] = pack2(t1[2 * c], t1[2 * c + 1]);
                w2[c] = pack2(t3[2 * c], t3[2 * c + 1]);
            }
            ab1[t] = __builtin_bit_cast(v16bf, w1);
            ab2[t] = __builtin_bit_cast(v16bf, w2);
        }
    }

    // 5 conv fields: mu1, mu2, E[xy], E[x^2], E[y^2]
    v8f res[5];
#pragma unroll
    for (int f = 0; f < 5; ++f) {
        v16bf A0, A1;
        if (f == 0)      { A0 = ab1[0];                  A1 = ab1[1]; }
        else if (f == 1) { A0 = ab2[0];                  A1 = ab2[1]; }
        else if (f == 2) { A0 = pkmul_bf16(ab1[0], ab2[0]); A1 = pkmul_bf16(ab1[1], ab2[1]); }
        else if (f == 3) { A0 = pkmul_bf16(ab1[0], ab1[0]); A1 = pkmul_bf16(ab1[1], ab1[1]); }
        else             { A0 = pkmul_bf16(ab2[0], ab2[0]); A1 = pkmul_bf16(ab2[1], ab2[1]); }

        v8f z = {};
        // Stage 1: horizontal pass, rows 0-15 and 16-31 of the halo region
        v8f d0 = __builtin_amdgcn_wmma_f32_16x16x32_bf16(
            false, A0, false, Hb, (short)0, z, false, false);
        v8f d1 = __builtin_amdgcn_wmma_f32_16x16x32_bf16(
            false, A1, false, Hb, (short)0, z, false, false);

        // Pack adjacent Mid rows (= adjacent K) into bf16 dwords, store to LDS.
        //   dword jj of d0 -> K-pair row kk = 4*hi + jj
        //   dword jj of d1 -> K-pair row kk = 8 + 4*hi + jj
#pragma unroll
        for (int jj = 0; jj < 4; ++jj) {
            midp[wave][(4 * hi + jj) * 20 + m]     = pack2(d0[2 * jj], d0[2 * jj + 1]);
            midp[wave][(8 + 4 * hi + jj) * 20 + m] = pack2(d1[2 * jj], d1[2 * jj + 1]);
        }
        // Wave-private scratch: same-wave DS ops are in-order in hardware;
        // compiler ordering is enforced by aliasing + this fence.
        asm volatile("" ::: "memory");

        // B-operand: elem i -> K = i + 16*hi, N = m  => dword c = K-pair 8*hi+c
        v8u uv;
#pragma unroll
        for (int c = 0; c < 8; ++c)
            uv[c] = midp[wave][(8 * hi + c) * 20 + m];
        v16bf Bm = __builtin_bit_cast(v16bf, uv);
        asm volatile("" ::: "memory");   // WAR guard vs next field's stores

        // Stage 2: vertical pass
        res[f] = __builtin_amdgcn_wmma_f32_16x16x32_bf16(
            false, Va, false, Bm, (short)0, z, false, false);
    }

    // Per-lane SSIM over its 8 output pixels (C/D layout)
    float lsum = 0.0f;
#pragma unroll
    for (int e = 0; e < 8; ++e) {
        const float mu1 = res[0][e];
        const float mu2 = res[1][e];
        const float m12 = mu1 * mu2;
        const float s12 = res[2][e] - m12;
        const float v1  = res[3][e] - mu1 * mu1;
        const float v2  = res[4][e] - mu2 * mu2;
        const float num = (2.0f * m12 + C1) * (2.0f * s12 + C2);
        const float den = (mu1 * mu1 + mu2 * mu2 + C1) * (v1 + v2 + C2);
        lsum += num / den;
    }

    // wave32 reduce
#pragma unroll
    for (int off = 16; off > 0; off >>= 1)
        lsum += __shfl_down(lsum, off, 32);
    if (lane == 0) wpart[wave] = lsum;
    __syncthreads();
    if (threadIdx.x == 0) {
        float bsum = 0.0f;
#pragma unroll
        for (int w = 0; w < 8; ++w) bsum += wpart[w];
        atomicAdd(&acc[blockIdx.x & 63], (double)bsum);
    }
}

__global__ void ssim_final_kernel(const double* acc, float* out) {
    if (threadIdx.x == 0) {
        double s = 0.0;
        for (int i = 0; i < 64; ++i) s += acc[i];
        out[0] = (float)(1.0 - s / N_PIX);
    }
}

extern "C" void kernel_launch(void* const* d_in, const int* in_sizes, int n_in,
                              void* d_out, int out_size, void* d_ws, size_t ws_size,
                              hipStream_t stream) {
    const float* img1 = (const float*)d_in[0];
    const float* img2 = (const float*)d_in[1];
    float* out  = (float*)d_out;
    double* acc = (double*)d_ws;     // 64 doubles = 512 B of scratch

    ssim_init_kernel<<<1, 64, 0, stream>>>(acc);
    // 32 images * 64*64 tiles = 131072 tiles, 8 tiles (waves) per 256-thread block
    ssim_main_kernel<<<131072 / 8, 256, 0, stream>>>(img1, img2, acc);
    ssim_final_kernel<<<1, 1, 0, stream>>>(acc, out);
}
